// KANLayer_22686017257519
// MI455X (gfx1250) — matrix operand order repaired and checked by reading, not compiled, not run on hardware
//
#include <hip/hip_runtime.h>
#include <stdint.h>

// ---------------------------------------------------------------------------
// KAN layer fused as one f16-WMMA GEMM on gfx1250 (wave32, WGP).
//   out[M=8192, N=1024] = X @ baseW^T  +  basis(X) @ splineW^T
// basis(X) expands K from 1024 -> 12288 (12 tent knots per feature; only 6
// unique values since edge knots repeat 4x), computed on the fly in packed
// f16 (v_pk_*). Both contractions accumulate into the same f32 WMMA
// accumulators. LDS is double-buffered (one barrier per chunk).
//
// Weight path (ASYNCW=true, needs 26MB workspace): weights are pre-converted
// to f16 once, then each chunk's B panel is streamed into LDS with
// global_load_async_to_lds_b128 (ASYNCcnt) -- zero VALU / zero ds_store on
// the B side. Fallback (ASYNCW=false): convert f32->f16 in-kernel.
// ---------------------------------------------------------------------------

typedef __attribute__((ext_vector_type(16))) _Float16 v16h;
typedef __attribute__((ext_vector_type(8)))  float    v8f;
typedef __attribute__((ext_vector_type(2)))  _Float16 h2;
typedef __attribute__((ext_vector_type(4)))  _Float16 h4;
typedef __attribute__((ext_vector_type(4)))  float    f4;

#define IN_F    1024
#define OUT_F   1024
#define N_ROWS  8192      // 4 * 2048
#define NB      12        // knots per feature
#define BM      128
#define BN      128
#define CI      8         // input features per spline chunk
#define BK      (CI*NB)   // 96 = 3 WMMA k-steps
#define SA      104       // LDS stride (halves): 208B/row -> bank-conflict free
#define SB      104
#define BASE_CK 32        // base-GEMM K chunk (1 WMMA k-step)

struct Frag32B { uint4 lo, hi; };
struct U2 { unsigned int a, b; };

// ---- gfx1250 async LDS helpers (inline asm; ASYNCcnt-tracked) -------------
__device__ __forceinline__ void async_ld16(const unsigned short* g,
                                           unsigned short* l) {
  // LDS dest address is taken from a VGPR (low 32 bits of generic pointer = LDS offset)
  asm volatile("global_load_async_to_lds_b128 %0, %1, off"
               :: "v"((unsigned)(size_t)l), "v"(g) : "memory");
}
__device__ __forceinline__ void async_wait0() {
  asm volatile("s_wait_asynccnt 0" ::: "memory");
}

// Two contiguous 16B LDS loads -> one v16h WMMA fragment (contiguous k = 16*khalf + s, same for A and B)
__device__ __forceinline__ v16h ld_frag(const unsigned short* p0,
                                        const unsigned short* p1) {
  Frag32B t;
  t.lo = *(const uint4*)p0;
  t.hi = *(const uint4*)p1;
  return __builtin_bit_cast(v16h, t);
}

// ---- A side: tent basis of x, packed f16 ----------------------------------
__device__ __forceinline__ void stage_splineA(const float* __restrict__ X,
                                              unsigned short* __restrict__ A,
                                              int row0, int i0, int tid) {
  const h2 kEdge  = {(_Float16)-1.0f, (_Float16)1.0f};
  const h2 kMid0  = {(_Float16)-0.6f, (_Float16)-0.2f};
  const h2 kMid1  = {(_Float16)0.2f,  (_Float16)0.6f};
  const h2 kScale = {(_Float16)2.5f,  (_Float16)2.5f};
  const h2 kOne   = {(_Float16)1.0f,  (_Float16)1.0f};
  const h2 kZero  = {(_Float16)0.0f,  (_Float16)0.0f};
#pragma unroll
  for (int p = 0; p < 4; ++p) {
    int idx = p * 256 + tid;                        // (row, f) pairs
    int r = idx >> 3, f = idx & 7;
    float xv = X[(size_t)(row0 + r) * IN_F + i0 + f];
    float xc = fminf(1.0f, fmaxf(-1.0f, xv));
    _Float16 xh = (_Float16)xc;
    h2 x2 = {xh, xh};
    // 6 unique tent values via 3 packed evaluations
    h2 e  = __builtin_elementwise_max(
        kOne - __builtin_elementwise_abs((x2 - kEdge) * kScale), kZero);
    h2 m0 = __builtin_elementwise_max(
        kOne - __builtin_elementwise_abs((x2 - kMid0) * kScale), kZero);
    h2 m1 = __builtin_elementwise_max(
        kOne - __builtin_elementwise_abs((x2 - kMid1) * kScale), kZero);
    h2 en = {e.x, e.x};
    h2 ep = {e.y, e.y};
    unsigned int un = __builtin_bit_cast(unsigned int, en);
    unsigned int up = __builtin_bit_cast(unsigned int, ep);
    unsigned int* dst = (unsigned int*)&A[r * SA + f * NB];
    dst[0] = un; dst[1] = un;                        // k=0..3  (knot -1 x4)
    dst[2] = __builtin_bit_cast(unsigned int, m0);   // k=4,5
    dst[3] = __builtin_bit_cast(unsigned int, m1);   // k=6,7
    dst[4] = up; dst[5] = up;                        // k=8..11 (knot +1 x4)
  }
}

// ---- B side, fallback: convert f32 weights in-kernel ----------------------
__device__ __forceinline__ void stage_splineB_cvt(const float* __restrict__ Wsp,
                                                  unsigned short* __restrict__ B,
                                                  int col0, int i0, int tid) {
#pragma unroll
  for (int p = 0; p < 4; ++p) {
    int idx = p * 256 + tid;                        // (n, f) pairs
    int n = idx >> 3, f = idx & 7;
    const f4* wp =
        (const f4*)(Wsp + ((size_t)(col0 + n) * IN_F + (i0 + f)) * NB);
    h4 w0 = __builtin_convertvector(wp[0], h4);
    h4 w1 = __builtin_convertvector(wp[1], h4);
    h4 w2 = __builtin_convertvector(wp[2], h4);
    unsigned int* dst = (unsigned int*)&B[n * SB + f * NB];
    U2 u;
    u = __builtin_bit_cast(U2, w0); dst[0] = u.a; dst[1] = u.b;
    u = __builtin_bit_cast(U2, w1); dst[2] = u.a; dst[3] = u.b;
    u = __builtin_bit_cast(U2, w2); dst[4] = u.a; dst[5] = u.b;
  }
}

// ---- B side, async: f16 weights DMA'd global -> LDS -----------------------
__device__ __forceinline__ void stage_splineB_async(
    const unsigned short* __restrict__ W16, unsigned short* __restrict__ B,
    int col0, int i0, int tid) {
  // panel row n: 96 contiguous halves = 12 x 16B chunks; 128*12 = 1536 chunks
#pragma unroll
  for (int p = 0; p < 6; ++p) {
    int q = p * 256 + tid;
    int n = q / 12, c = q - n * 12;
    const unsigned short* g =
        W16 + ((size_t)(col0 + n) * IN_F + i0) * NB + c * 8;
    async_ld16(g, &B[n * SB + c * 8]);
  }
}

// ---- base chunk staging ----------------------------------------------------
__device__ __forceinline__ void stage_baseA(const float* __restrict__ X,
                                            unsigned short* __restrict__ A,
                                            int row0, int i0, int tid) {
#pragma unroll
  for (int p = 0; p < 2; ++p) {
    int idx = p * 256 + tid;                        // (row, k/4) quads
    int r = idx >> 3, k = (idx & 7) * 4;
    f4 xa = *(const f4*)&X[(size_t)(row0 + r) * IN_F + i0 + k];
    *(unsigned long long*)&A[r * SA + k] =
        __builtin_bit_cast(unsigned long long, __builtin_convertvector(xa, h4));
    // second half of rows
    r += 64;
    f4 xb = *(const f4*)&X[(size_t)(row0 + r) * IN_F + i0 + k];
    *(unsigned long long*)&A[r * SA + k] =
        __builtin_bit_cast(unsigned long long, __builtin_convertvector(xb, h4));
  }
}
__device__ __forceinline__ void stage_baseB_cvt(const float* __restrict__ Wb,
                                                unsigned short* __restrict__ B,
                                                int col0, int i0, int tid) {
#pragma unroll
  for (int p = 0; p < 4; ++p) {
    int idx = p * 256 + tid;
    int n = idx >> 3, k = (idx & 7) * 4;
    f4 wa = *(const f4*)&Wb[(size_t)(col0 + n) * IN_F + i0 + k];
    *(unsigned long long*)&B[n * SB + k] =
        __builtin_bit_cast(unsigned long long, __builtin_convertvector(wa, h4));
  }
}
__device__ __forceinline__ void stage_baseB_async(
    const unsigned short* __restrict__ Wb16, unsigned short* __restrict__ B,
    int col0, int i0, int tid) {
  // row n: 32 halves = 4 x 16B chunks; 128*4 = 512 chunks
#pragma unroll
  for (int p = 0; p < 2; ++p) {
    int q = p * 256 + tid;
    int n = q >> 2, c = q & 3;
    const unsigned short* g = Wb16 + (size_t)(col0 + n) * IN_F + i0 + c * 8;
    async_ld16(g, &B[n * SB + c * 8]);
  }
}

// NSTEPS x (K=32) WMMA steps over the wave's 2x4 grid of 16x16 tiles
template <int NSTEPS>
__device__ __forceinline__ void mma_block(const unsigned short* __restrict__ A,
                                          const unsigned short* __restrict__ B,
                                          v8f (&acc)[2][4], int wm, int wn,
                                          int lrow, int khalf) {
#pragma unroll
  for (int ks = 0; ks < NSTEPS; ++ks) {
    const int k0 = ks * 32;
    v16h a[2], b[4];
#pragma unroll
    for (int sm = 0; sm < 2; ++sm) {
      const unsigned short* ar = &A[(wm * 32 + sm * 16 + lrow) * SA + k0];
      a[sm] = ld_frag(ar + 16 * khalf, ar + 16 * khalf + 8);
    }
#pragma unroll
    for (int sn = 0; sn < 4; ++sn) {
      const unsigned short* br =
          &B[(wn * 64 + sn * 16 + lrow) * SB + k0 + 16 * khalf];
      b[sn] = ld_frag(br, br + 8);
    }
#pragma unroll
    for (int sm = 0; sm < 2; ++sm)
#pragma unroll
      for (int sn = 0; sn < 4; ++sn)
        acc[sm][sn] = __builtin_amdgcn_wmma_f32_16x16x32_f16(
            false, a[sm], false, b[sn], (short)0, acc[sm][sn], false, false);
  }
}

// ---- one-shot f32 -> f16 weight image into workspace -----------------------
__global__ void convert_weights_f16(const float* __restrict__ Wsp,
                                    const float* __restrict__ Wb,
                                    unsigned short* __restrict__ W16) {
  const size_t NSPQ = (size_t)OUT_F * IN_F * NB / 4;   // spline quads
  const size_t NBQ  = (size_t)OUT_F * IN_F / 4;        // base quads
  size_t q = (size_t)blockIdx.x * blockDim.x + threadIdx.x;
  const float* src;
  unsigned short* dst;
  if (q < NSPQ) {
    src = Wsp + q * 4;
    dst = W16 + q * 4;
  } else if (q < NSPQ + NBQ) {
    size_t r = q - NSPQ;
    src = Wb + r * 4;
    dst = W16 + NSPQ * 4 + r * 4;
  } else {
    return;
  }
  f4 v = *(const f4*)src;
  *(unsigned long long*)dst =
      __builtin_bit_cast(unsigned long long, __builtin_convertvector(v, h4));
}

template <bool ASYNCW>
__global__ __launch_bounds__(256)
void kan_fused_wmma(const float* __restrict__ X,            // (8192, 1024)
                    const float* __restrict__ Wsp,          // (1024,1024,12)
                    const float* __restrict__ Wb,           // (1024, 1024)
                    const unsigned short* __restrict__ Wsp16,
                    const unsigned short* __restrict__ Wb16,
                    float* __restrict__ Out) {              // (8192, 1024)
  __shared__ __align__(16) unsigned short lsA[2][BM * SA];
  __shared__ __align__(16) unsigned short lsB[2][BN * SB];

  const int tid   = threadIdx.x;
  const int lane  = tid & 31;
  const int wid   = tid >> 5;
  const int wm    = wid & 3;          // 4 waves along M (32 rows each)
  const int wn    = wid >> 2;         // 2 waves along N (64 cols each)
  const int lrow  = lane & 15;
  const int khalf = lane >> 4;        // K-half select (lanes 16-31)
  const int mTile = blockIdx.x >> 3;  // 64 tiles of 128 rows
  const int nTile = blockIdx.x & 7;   // 8 tiles of 128 cols
  const int row0  = mTile * BM;
  const int col0  = nTile * BN;

  v8f acc[2][4] = {};                 // 32x64 f32 per wave

  // ---------------- spline contraction: K = 1024 * 12 --------------------
  const int NC = IN_F / CI;           // 128 chunks
  stage_splineA(X, lsA[0], row0, 0, tid);
  if constexpr (ASYNCW) {
    stage_splineB_async(Wsp16, lsB[0], col0, 0, tid);
    async_wait0();
  } else {
    stage_splineB_cvt(Wsp, lsB[0], col0, 0, tid);
  }
  __syncthreads();
  for (int ic = 0; ic < NC; ++ic) {
    const int cur = ic & 1;
    if (ic + 1 < NC) {                // stage next chunk into the other buffer
      stage_splineA(X, lsA[cur ^ 1], row0, (ic + 1) * CI, tid);
      if constexpr (ASYNCW)
        stage_splineB_async(Wsp16, lsB[cur ^ 1], col0, (ic + 1) * CI, tid);
      else
        stage_splineB_cvt(Wsp, lsB[cur ^ 1], col0, (ic + 1) * CI, tid);
    }
    mma_block<3>(lsA[cur], lsB[cur], acc, wm, wn, lrow, khalf);
    if constexpr (ASYNCW) async_wait0();
    __syncthreads();
  }

  // ---------------- base contraction: K = 1024, same accumulators --------
  const int NCB = IN_F / BASE_CK;     // 32 chunks
  stage_baseA(X, lsA[0], row0, 0, tid);
  if constexpr (ASYNCW) {
    stage_baseB_async(Wb16, lsB[0], col0, 0, tid);
    async_wait0();
  } else {
    stage_baseB_cvt(Wb, lsB[0], col0, 0, tid);
  }
  __syncthreads();
  for (int ic = 0; ic < NCB; ++ic) {
    const int cur = ic & 1;
    if (ic + 1 < NCB) {
      stage_baseA(X, lsA[cur ^ 1], row0, (ic + 1) * BASE_CK, tid);
      if constexpr (ASYNCW)
        stage_baseB_async(Wb16, lsB[cur ^ 1], col0, (ic + 1) * BASE_CK, tid);
      else
        stage_baseB_cvt(Wb, lsB[cur ^ 1], col0, (ic + 1) * BASE_CK, tid);
    }
    mma_block<1>(lsA[cur], lsB[cur], acc, wm, wn, lrow, khalf);
    if constexpr (ASYNCW) async_wait0();
    __syncthreads();
  }

  // ---------------- store D tiles (lane=col, VGPR j = row j + 8*khalf) ----
#pragma unroll
  for (int sm = 0; sm < 2; ++sm) {
#pragma unroll
    for (int sn = 0; sn < 4; ++sn) {
      const int col   = col0 + wn * 64 + sn * 16 + lrow;
      const int rbase = row0 + wm * 32 + sm * 16 + 8 * khalf;
#pragma unroll
      for (int j = 0; j < 8; ++j)
        Out[(size_t)(rbase + j) * OUT_F + col] = acc[sm][sn][j];
    }
  }
}

extern "C" void kernel_launch(void* const* d_in, const int* in_sizes, int n_in,
                              void* d_out, int out_size, void* d_ws,
                              size_t ws_size, hipStream_t stream) {
  const float* x   = (const float*)d_in[0];   // (4,2048,1024) f32
  const float* wsp = (const float*)d_in[1];   // (1024,1024,12) f32
  const float* wb  = (const float*)d_in[2];   // (1024,1024) f32
  float*       out = (float*)d_out;           // (4,2048,1024) f32

  const size_t NSP  = (size_t)OUT_F * IN_F * NB;        // 12,582,912 halves
  const size_t NBW  = (size_t)OUT_F * IN_F;             //  1,048,576 halves
  const size_t need = (NSP + NBW) * sizeof(unsigned short);  // ~26 MB
  const int grid = (N_ROWS / BM) * (OUT_F / BN);        // 512 blocks

  if (d_ws != nullptr && ws_size >= need) {
    unsigned short* w16 = (unsigned short*)d_ws;
    const size_t quads = (NSP + NBW) / 4;
    convert_weights_f16<<<(int)((quads + 255) / 256), 256, 0, stream>>>(
        wsp, wb, w16);
    kan_fused_wmma<true><<<grid, 256, 0, stream>>>(x, wsp, wb, w16, w16 + NSP,
                                                   out);
  } else {
    kan_fused_wmma<false><<<grid, 256, 0, stream>>>(x, wsp, wb, nullptr,
                                                    nullptr, out);
  }
}
